// PointUpsample_subpixel_30176440222308
// MI455X (gfx1250) — matrix operand order repaired and checked by reading, not compiled, not run on hardware
//
#include <hip/hip_runtime.h>
#include <math.h>

typedef __bf16 bh16;
typedef __attribute__((ext_vector_type(4)))  float  v4f;
typedef __attribute__((ext_vector_type(8)))  float  v8f;
typedef __attribute__((ext_vector_type(4)))  __bf16 v4bf;
typedef __attribute__((ext_vector_type(8)))  __bf16 v8bf;
typedef __attribute__((ext_vector_type(16))) __bf16 v16bf;

#define BL    65536   // B*L
#define CIN   256
#define PE    96
#define RPE   90
#define KDIM  448     // CIN + 2*PE
#define NOUT  1024
#define MT    64      // rows per workgroup
#define NT    256     // output cols per workgroup
#define STR   456     // LDS row stride in bf16 elems (456*2B row offset -> distinct banks)

// ---- fp32 -> (hi,lo) bf16 split: hi = bit-truncation (exact), lo = RNE(residual) ----
__device__ __forceinline__ unsigned short f32_to_bf16_rne(float f) {
    unsigned u = __float_as_uint(f);
    unsigned r = u + 0x7FFFu + ((u >> 16) & 1u);
    return (unsigned short)(r >> 16);
}
__device__ __forceinline__ void split1(float v, bh16* hi, bh16* lo) {
    unsigned b = __float_as_uint(v);
    *hi = __builtin_bit_cast(bh16, (unsigned short)(b >> 16));
    float resid = v - __uint_as_float(b & 0xFFFF0000u);   // exact
    *lo = __builtin_bit_cast(bh16, f32_to_bf16_rne(resid));
}
__device__ __forceinline__ float bf16_to_f32(bh16 h) {
    unsigned u = (unsigned)__builtin_bit_cast(unsigned short, h) << 16;
    return __uint_as_float(u);
}
__device__ __forceinline__ v16bf cat8(v8bf a, v8bf b) {
    return __builtin_shufflevector(a, b, 0,1,2,3,4,5,6,7,8,9,10,11,12,13,14,15);
}
#define WMMA_BF16(A, B, C) \
    __builtin_amdgcn_wmma_f32_16x16x32_bf16(false, (A), false, (B), (short)0, (C), false, false)

// ---- prolog: decompose sub_w (1024x448 fp32) into bf16 hi/lo planes in workspace ----
__global__ __launch_bounds__(256) void decompose_w_kernel(
    const float* __restrict__ w, bh16* __restrict__ whi, bh16* __restrict__ wlo)
{
    int i = blockIdx.x * 256 + threadIdx.x;
    if (i < NOUT * KDIM) {
        bh16 h, l;
        split1(w[i], &h, &l);
        whi[i] = h;
        wlo[i] = l;
    }
}

// ---- fused: build x_pe tile (bf16 hi/lo in LDS) + bf16x3 WMMA GEMM + epilogue ----
__global__ __launch_bounds__(256) void fused_upsample_kernel(
    const float* __restrict__ x, const float* __restrict__ pc,
    const float* __restrict__ lff_w, const float* __restrict__ lff_b,
    const float* __restrict__ const_pe,
    const bh16* __restrict__ whi, const bh16* __restrict__ wlo,
    const float* __restrict__ sub_b, float* __restrict__ xup,
    float* __restrict__ pcup)
{
    __shared__ bh16  s_ahi[MT][STR];
    __shared__ bh16  s_alo[MT][STR];
    __shared__ float s_pc[MT][3];

    const int tid = threadIdx.x;
    const int mb0 = blockIdx.x * MT;          // first global row of this block

    // stage pc[64][3]
    if (tid < MT * 3) {
        s_pc[tid / 3][tid % 3] = pc[(size_t)(mb0 + tid / 3) * 3 + (tid % 3)];
    }
    // x tile -> cols [0,256) as bf16 hi/lo : 64 rows x 64 float4
    for (int i = tid; i < MT * (CIN / 4); i += 256) {
        int r = i >> 6, c4 = i & 63;
        v4f xv = *(const v4f*)(x + (size_t)(mb0 + r) * CIN + c4 * 4);
        v4bf h4, l4;
#pragma unroll
        for (int j = 0; j < 4; ++j) {
            bh16 h, l;
            split1(xv[j], &h, &l);
            h4[j] = h; l4[j] = l;
        }
        *(v4bf*)(&s_ahi[r][c4 * 4]) = h4;
        *(v4bf*)(&s_alo[r][c4 * 4]) = l4;
    }
    __syncthreads();

    // pc_up passthrough (only the up==0 grid slice writes it): 64*4*3 = 768
    if (blockIdx.y == 0) {
        for (int i = tid; i < MT * 4 * 3; i += 256) {
            int r = i / 12, rem = i % 12, u = rem / 3, c = rem % 3;
            pcup[(size_t)((mb0 + r) * 4 + u) * 3 + c] = s_pc[r][c];
        }
    }

    // lpe = sin(LFF(pc)) -> cols [256,352)
    for (int i = tid; i < MT * PE; i += 256) {
        int r = i / PE, d = i % PE;
        float a = s_pc[r][0] * lff_w[d * 3 + 0] + s_pc[r][1] * lff_w[d * 3 + 1] +
                  s_pc[r][2] * lff_w[d * 3 + 2] + lff_b[d];
        bh16 h, l;
        split1(sinf(a), &h, &l);
        s_ahi[r][CIN + d] = h;
        s_alo[r][CIN + d] = l;
    }
    // cpe: bilinear gather from const_pe[3][96][90] -> cols [352,448)
    for (int i = tid; i < MT * PE; i += 256) {
        int r = i / PE, d = i % PE;
        float sum = 0.f;
#pragma unroll
        for (int ax = 0; ax < 3; ++ax) {
            float p  = s_pc[r][ax];
            float ix = ((p + 1.f) * (float)RPE - 1.f) * 0.5f;
            float fi = floorf(ix);
            float w  = ix - fi;
            int  i0  = (int)fi, i1 = i0 + 1;
            const float* row = const_pe + (size_t)(ax * PE + d) * RPE;
            float v0 = (i0 >= 0 && i0 < RPE) ? row[i0] : 0.f;
            float v1 = (i1 >= 0 && i1 < RPE) ? row[i1] : 0.f;
            sum += v0 * (1.f - w) + v1 * w;
        }
        bh16 h, l;
        split1(sum * 0.57735026919f, &h, &l);
        s_ahi[r][CIN + PE + d] = h;
        s_alo[r][CIN + PE + d] = l;
    }
    __syncthreads();

    // ---- bf16x3 GEMM: 8 waves = 4(M) x 2(N); each wave 16 rows x 128 cols ----
    const int lane = tid & 31;
    const int wv   = tid >> 5;
    const int hf   = lane >> 4;       // K-half selector per ISA 16-bit A/B layouts
    const int l16  = lane & 15;       // A: M row, B/D: N col
    const int mq   = wv & 3;          // M sub-tile 0..3
    const int nq   = wv >> 1 >> 1;    // N sub-slab 0..1  (wv>>2)

    const bh16* ahp = &s_ahi[mq * 16 + l16][8 * hf];   // runs at kb and kb+16
    const bh16* alp = &s_alo[mq * 16 + l16][8 * hf];

    const int ncol0 = blockIdx.y * NT + nq * 128 + l16;
    const bh16* bhp = whi + (size_t)ncol0 * KDIM + 8 * hf;
    const bh16* blp = wlo + (size_t)ncol0 * KDIM + 8 * hf;

    v8f acc[8];
#pragma unroll
    for (int t = 0; t < 8; ++t)
        acc[t] = (v8f){0.f, 0.f, 0.f, 0.f, 0.f, 0.f, 0.f, 0.f};

    for (int kc = 0; kc < KDIM / 32; ++kc) {
        const int kb = kc * 32;
        v16bf ahi = cat8(*(const v8bf*)(ahp + kb), *(const v8bf*)(ahp + kb + 16));
        v16bf alo = cat8(*(const v8bf*)(alp + kb), *(const v8bf*)(alp + kb + 16));
#pragma unroll
        for (int t = 0; t < 8; ++t) {
            const bh16* bh = bhp + t * 16 * KDIM + kb;
            const bh16* bl = blp + t * 16 * KDIM + kb;
            v16bf bhi = cat8(*(const v8bf*)(bh), *(const v8bf*)(bh + 16));
            v16bf blo = cat8(*(const v8bf*)(bl), *(const v8bf*)(bl + 16));
            acc[t] = WMMA_BF16(ahi, bhi, acc[t]);
            acc[t] = WMMA_BF16(ahi, blo, acc[t]);
            acc[t] = WMMA_BF16(alo, bhi, acc[t]);
        }
    }

    // ---- epilogue: gain, bias, lrelu*sqrt2, +residual, /sqrt2, store ----
    const float kGain = 0.047245559f;   // 1/sqrt(448)
    const int   up    = blockIdx.y;
#pragma unroll
    for (int t = 0; t < 8; ++t) {
        int   ocw  = nq * 128 + t * 16 + l16;   // 0..255 within OUT
        float bias = sub_b[blockIdx.y * NT + ocw];
#pragma unroll
        for (int v = 0; v < 8; ++v) {
            int   rL = mq * 16 + v + 8 * hf;    // LDS row (D-matrix row layout)
            int   m  = mb0 + rL;                // global row over B*L
            float y  = acc[t][v] * kGain + bias;
            y = (y >= 0.f ? y : 0.2f * y) * 1.41421356237f;
            float res = bf16_to_f32(s_ahi[rL][ocw]) + bf16_to_f32(s_alo[rL][ocw]);
            xup[(size_t)(m * 4 + up) * 256 + ocw] = (res + y) * 0.70710678119f;
        }
    }
}

extern "C" void kernel_launch(void* const* d_in, const int* in_sizes, int n_in,
                              void* d_out, int out_size, void* d_ws, size_t ws_size,
                              hipStream_t stream) {
    const float* x      = (const float*)d_in[0];
    const float* pc     = (const float*)d_in[1];
    const float* lff_w  = (const float*)d_in[2];
    const float* lff_b  = (const float*)d_in[3];
    const float* cpet   = (const float*)d_in[4];
    const float* sub_w  = (const float*)d_in[5];
    const float* sub_b  = (const float*)d_in[6];

    float* xup  = (float*)d_out;
    float* pcup = xup + (size_t)8 * 32768 * 256;     // pc_up follows x_up

    bh16* whi = (bh16*)d_ws;                          // 1024*448 bf16
    bh16* wlo = whi + (size_t)NOUT * KDIM;            // 1024*448 bf16 (1.75 MB total)

    decompose_w_kernel<<<dim3((NOUT * KDIM + 255) / 256), 256, 0, stream>>>(sub_w, whi, wlo);

    fused_upsample_kernel<<<dim3(BL / MT, NOUT / NT), 256, 0, stream>>>(
        x, pc, lff_w, lff_b, cpet, whi, wlo, sub_b, xup, pcup);
}
